// OnetPlusPlus_24077586661645
// MI455X (gfx1250) — matrix operand order repaired and checked by reading, not compiled, not run on hardware
//
#include <hip/hip_runtime.h>

typedef __attribute__((ext_vector_type(16))) _Float16 v16h;
typedef __attribute__((ext_vector_type(8)))  _Float16 v8h;
typedef __attribute__((ext_vector_type(4)))  _Float16 v4h;
typedef __attribute__((ext_vector_type(8)))  float    v8f;

#define LEAKY 0.2f
__device__ __forceinline__ float lrelu(float x) {
    return fmaxf(x, 0.f) + LEAKY * fminf(x, 0.f);
}

// ---------------------------------------------------------------------------
// Pack a row-major f32 [K,N] weight matrix into WMMA-B fragment order (f16).
// B-frag (16-bit, 32x16): lane L, elem i -> K = kk*32 + (i&7) + 8*(L>>4) + 16*(i>>3),
//                                          N = ntile*16 + (L&15)
// dst[( (ntile*KK + kk)*32 + lane )*16 + i]  (== linear tid)
// ---------------------------------------------------------------------------
__global__ void packw_kernel(const float* __restrict__ src, _Float16* __restrict__ dst,
                             int K, int N) {
    int tid = blockIdx.x * 256 + threadIdx.x;
    if (tid >= K * N) return;
    int i    = tid & 15;
    int lane = (tid >> 4) & 31;
    int frag = tid >> 9;          // ntile*KK + kk
    int KK   = K >> 5;
    int kk   = frag % KK;
    int nt   = frag / KK;
    int k = kk * 32 + (i & 7) + 8 * (lane >> 4) + 16 * (i >> 3);
    int n = nt * 16 + (lane & 15);
    dst[tid] = (_Float16)src[(size_t)k * N + n];
}

// ---------------------------------------------------------------------------
// KNN: per-batch block; top-16 nearest of 4096 points to the query.
// ---------------------------------------------------------------------------
__global__ void knn_kernel(const float* __restrict__ gpc, const float* __restrict__ query,
                           float* __restrict__ nnpts, float* __restrict__ knnpos) {
    __shared__ unsigned long long keys[4096];
    __shared__ unsigned long long red[256];
    __shared__ int sel[16];
    int b = blockIdx.x, t = threadIdx.x;
    float qx = query[b * 3 + 0], qy = query[b * 3 + 1], qz = query[b * 3 + 2];
    const float* P = gpc + (size_t)b * 4096 * 3;
    for (int p = t; p < 4096; p += 256) {
        float dx = qx - P[p * 3 + 0];
        float dy = qy - P[p * 3 + 1];
        float dz = qz - P[p * 3 + 2];
        float d2 = dx * dx + dy * dy + dz * dz;   // >= 0 -> bit order == float order
        keys[p] = ((unsigned long long)__float_as_uint(d2) << 32) | (unsigned)p;
    }
    __syncthreads();
    for (int pass = 0; pass < 16; ++pass) {
        unsigned long long m = ~0ull;
        for (int p = t; p < 4096; p += 256) {
            unsigned long long k = keys[p];
            m = k < m ? k : m;
        }
        red[t] = m;
        __syncthreads();
        for (int s = 128; s > 0; s >>= 1) {
            if (t < s) { if (red[t + s] < red[t]) red[t] = red[t + s]; }
            __syncthreads();
        }
        if (t == 0) {
            int id = (int)(red[0] & 0xffffffffu);
            sel[pass] = id;
            keys[id] = ~0ull;
        }
        __syncthreads();
    }
    if (t < 16) {
        int id = sel[t];
        float nx = P[id * 3 + 0], ny = P[id * 3 + 1], nz = P[id * 3 + 2];
        int o = (b * 16 + t) * 3;
        nnpts[o] = nx;       nnpts[o + 1] = ny;       nnpts[o + 2] = nz;
        knnpos[o] = qx - nx; knnpos[o + 1] = qy - ny; knnpos[o + 2] = qz - nz;
    }
}

// ---------------------------------------------------------------------------
// Attention branch: recompute point_feature at the 16 KNN points (the only
// live use of the global encoder), attention over K=16, query pos-embedding.
// One block (128 threads) per batch; thread == channel.
// ---------------------------------------------------------------------------
__global__ void att_kernel(const float* __restrict__ nnpts, const float* __restrict__ knnpos,
                           const float* __restrict__ query,
                           const float* __restrict__ gew1, const float* __restrict__ geb1,
                           const float* __restrict__ gew2, const float* __restrict__ geb2,
                           const float* __restrict__ posw, const float* __restrict__ posb,
                           const float* __restrict__ gamma, const float* __restrict__ beta,
                           const float* __restrict__ apw, const float* __restrict__ apb,
                           const float* __restrict__ asw, const float* __restrict__ asb,
                           float* __restrict__ feat) {
    __shared__ float h[16][64];
    __shared__ float hq[16][128];
    int b = blockIdx.x, t = threadIdx.x;
    for (int e = t; e < 1024; e += 128) {
        int p = e >> 6, k = e & 63;
        const float* np = nnpts + (b * 16 + p) * 3;
        float v = np[0] * gew1[k] + np[1] * gew1[64 + k] + np[2] * gew1[128 + k] + geb1[k];
        h[p][k] = lrelu(v);
    }
    __syncthreads();
    int c = t;
    for (int p = 0; p < 16; ++p) {
        float acc = geb2[c];
        for (int k = 0; k < 64; ++k) acc += h[p][k] * gew2[k * 128 + c];
        float kf = lrelu(acc);                                   // knn_feature
        const float* kp = knnpos + (b * 16 + p) * 3;
        float pe = kp[0] * apw[c] + kp[1] * apw[128 + c] + kp[2] * apw[256 + c] + apb[c];
        hq[p][c] = kf + pe;
    }
    __syncthreads();
    float s[16];
    float mx = -3.4e38f;
    for (int p = 0; p < 16; ++p) {
        float acc = asb[c];
        for (int k = 0; k < 128; ++k) acc += hq[p][k] * asw[k * 128 + c];
        s[p] = acc;
        mx = fmaxf(mx, acc);
    }
    float sum = 0.f;
    for (int p = 0; p < 16; ++p) { s[p] = __expf(s[p] - mx); sum += s[p]; }
    float inv = 1.f / sum;
    float qf = 0.f;
    for (int p = 0; p < 16; ++p) qf += (s[p] * inv) * hq[p][c];
    const float* q = query + b * 3;
    float pv = q[0] * posw[c] + q[1] * posw[128 + c] + q[2] * posw[256 + c] + posb[c];
    float pfq = lrelu(gamma[c] * pv + beta[c]);
    feat[(size_t)b * 4608 + 2304 + c] = qf;
    feat[(size_t)b * 4608 + 2432 + c] = pfq;
}

// ---------------------------------------------------------------------------
// Local encoder (dominant GEMM): stage1 (3->64) into A-fragment registers,
// stage2 (64->256) via v_wmma_f32_16x16x32_f16, lrelu+max over points.
// Block = 256 thr (8 waves x 16 points) -> partial max [block][256].
// A-frag (16-bit, 16x32): lane L, elem i -> row M=L&15, K=i+8*((i>>3)+(L>>4)).
// ---------------------------------------------------------------------------
__global__ void le_kernel(const float* __restrict__ lpc,
                          const float* __restrict__ w1, const float* __restrict__ bias1,
                          const _Float16* __restrict__ w2p, const float* __restrict__ bias2,
                          float* __restrict__ partial) {
    __shared__ float sw1[192];
    __shared__ float sb1[64];
    __shared__ float wavemax[16][256];
    int t = threadIdx.x;
    int wave = t >> 5, lane = t & 31;
    int g = lane >> 4, col = lane & 15;
    if (t < 192) sw1[t] = w1[t];
    if (t < 64)  sb1[t] = bias1[t];
    __syncthreads();

    int b = blockIdx.x >> 3;
    int pt = (blockIdx.x & 7) * 128 + wave * 16 + col;
    const float* P = lpc + ((size_t)b * 1024 + pt) * 3;
    float x0 = P[0], x1 = P[1], x2 = P[2];

    v16h a[2];
#pragma unroll
    for (int kk = 0; kk < 2; ++kk) {
#pragma unroll
        for (int i = 0; i < 16; ++i) {
            int k = kk * 32 + i + 8 * ((i >> 3) + g);
            float hv = x0 * sw1[k] + x1 * sw1[64 + k] + x2 * sw1[128 + k] + sb1[k];
            a[kk][i] = (_Float16)lrelu(hv);
        }
    }

    const v16h* bw = (const v16h*)w2p;
#pragma unroll 1
    for (int nt = 0; nt < 16; ++nt) {
        v8f c = {};
        v16h bf0 = bw[(nt * 2 + 0) * 32 + lane];
        v16h bf1 = bw[(nt * 2 + 1) * 32 + lane];
        c = __builtin_amdgcn_wmma_f32_16x16x32_f16(false, a[0], false, bf0, (short)0, c, false, false);
        c = __builtin_amdgcn_wmma_f32_16x16x32_f16(false, a[1], false, bf1, (short)0, c, false, false);
        int n = nt * 16 + col;
        float bs = bias2[n];
        float m = -3.4e38f;
#pragma unroll
        for (int v = 0; v < 8; ++v) m = fmaxf(m, lrelu(c[v] + bs));   // rows M=v+8g
        wavemax[wave * 2 + g][n] = m;
    }
    __syncthreads();
    float m = wavemax[0][t];
#pragma unroll
    for (int r = 1; r < 16; ++r) m = fmaxf(m, wavemax[r][t]);
    partial[(size_t)blockIdx.x * 256 + t] = m;
}

__global__ void lered_kernel(const float* __restrict__ partial, float* __restrict__ feat) {
    int b = blockIdx.x, t = threadIdx.x;
    float m = partial[(size_t)(b * 8) * 256 + t];
    for (int c = 1; c < 8; ++c) m = fmaxf(m, partial[(size_t)(b * 8 + c) * 256 + t]);
    feat[(size_t)b * 4608 + 2048 + t] = m;   // local_c
}

// ---------------------------------------------------------------------------
// Voxel branches: one block per (batch, branch); two stride-2 SAME convs.
// ---------------------------------------------------------------------------
__global__ void vox_kernel(const float* __restrict__ pcv, const float* __restrict__ vox,
                           const float* __restrict__ gw1, const float* __restrict__ gb1,
                           const float* __restrict__ gw2, const float* __restrict__ gb2,
                           const float* __restrict__ fw1, const float* __restrict__ fb1,
                           const float* __restrict__ fw2, const float* __restrict__ fb2,
                           float* __restrict__ feat) {
    __shared__ float in[4096];
    __shared__ float o1[8192];
    __shared__ float sw1[432];
    __shared__ float sb1[16];
    int b = blockIdx.x >> 1, br = blockIdx.x & 1;
    const float* src = (br ? vox : pcv) + (size_t)b * 4096;
    const float* w1 = br ? fw1 : gw1;
    const float* b1 = br ? fb1 : gb1;
    const float* w2 = br ? fw2 : gw2;
    const float* b2 = br ? fb2 : gb2;
    int off = br ? 2560 : 0;
    int t = threadIdx.x;
    for (int e = t; e < 4096; e += 256) in[e] = src[e];
    for (int e = t; e < 432; e += 256) sw1[e] = w1[e];
    if (t < 16) sb1[t] = b1[t];
    __syncthreads();
    // conv1: 1 -> 16, 16^3 -> 8^3, SAME (pad_lo=0)
    for (int e = t; e < 8192; e += 256) {
        int c = e & 15, ow = (e >> 4) & 7, oh = (e >> 7) & 7, od = e >> 10;
        float acc = sb1[c];
        for (int kd = 0; kd < 3; ++kd) {
            int id = 2 * od + kd; if (id >= 16) continue;
            for (int kh = 0; kh < 3; ++kh) {
                int ih = 2 * oh + kh; if (ih >= 16) continue;
                for (int kw = 0; kw < 3; ++kw) {
                    int iw = 2 * ow + kw; if (iw >= 16) continue;
                    acc += in[id * 256 + ih * 16 + iw] * sw1[((kd * 3 + kh) * 3 + kw) * 16 + c];
                }
            }
        }
        o1[e] = lrelu(acc);
    }
    __syncthreads();
    // conv2: 16 -> 32, 8^3 -> 4^3
    for (int e = t; e < 2048; e += 256) {
        int c = e & 31, ow = (e >> 5) & 3, oh = (e >> 7) & 3, od = e >> 9;
        float acc = b2[c];
        for (int kd = 0; kd < 3; ++kd) {
            int id = 2 * od + kd; if (id >= 8) continue;
            for (int kh = 0; kh < 3; ++kh) {
                int ih = 2 * oh + kh; if (ih >= 8) continue;
                for (int kw = 0; kw < 3; ++kw) {
                    int iw = 2 * ow + kw; if (iw >= 8) continue;
                    const float* wv = w2 + (size_t)(((kd * 3 + kh) * 3 + kw) * 16) * 32 + c;
                    const float* iv = &o1[((id * 8 + ih) * 8 + iw) * 16];
                    for (int ci = 0; ci < 16; ++ci) acc += iv[ci] * wv[ci * 32];
                }
            }
        }
        // NCDHW reshape: index = c*64 + od*16 + oh*4 + ow
        feat[(size_t)b * 4608 + off + c * 64 + od * 16 + oh * 4 + ow] = lrelu(acc);
    }
}

// ---------------------------------------------------------------------------
// feat f32 -> f16 (decoder A operand)
// ---------------------------------------------------------------------------
__global__ void featcvt_kernel(const float* __restrict__ f, _Float16* __restrict__ h) {
    int base = (blockIdx.x * 256 + threadIdx.x) * 4;   // total 589824 elems
    float4 v = *(const float4*)(f + base);
    v4h o;
    o[0] = (_Float16)v.x; o[1] = (_Float16)v.y; o[2] = (_Float16)v.z; o[3] = (_Float16)v.w;
    *(v4h*)(h + base) = o;
}

// ---------------------------------------------------------------------------
// Decoder GEMM: [128,4608] x [4608,512] via WMMA. 32 blocks = N-tiles,
// 8 waves = M-tiles (16 batches). A frags: two aligned b128 loads from
// row-major f16 feat (k = kk*32 + 8g + {0..7, 16..23}).
// ---------------------------------------------------------------------------
__global__ void dec1_kernel(const _Float16* __restrict__ featH, const _Float16* __restrict__ w1p,
                            const float* __restrict__ bias1, float* __restrict__ hidden) {
    int t = threadIdx.x;
    int wave = t >> 5, lane = t & 31, g = lane >> 4, col = lane & 15;
    int nt = blockIdx.x;
    const _Float16* arow = featH + (size_t)(wave * 16 + col) * 4608;  // A row M = col
    const v16h* bw = (const v16h*)w1p;
    size_t bbase = (size_t)nt * 144 * 32 + lane;
    v8f c = {};
#pragma unroll 4
    for (int kk = 0; kk < 144; ++kk) {
        int k0 = kk * 32 + 8 * g;
        v8h alo = *(const v8h*)(arow + k0);
        v8h ahi = *(const v8h*)(arow + k0 + 16);
        v16h a = __builtin_shufflevector(alo, ahi, 0, 1, 2, 3, 4, 5, 6, 7,
                                         8, 9, 10, 11, 12, 13, 14, 15);
        v16h bf = bw[bbase + (size_t)kk * 32];
        if (kk + 8 < 144) __builtin_prefetch(&bw[bbase + (size_t)(kk + 8) * 32], 0, 1);
        c = __builtin_amdgcn_wmma_f32_16x16x32_f16(false, a, false, bf, (short)0, c, false, false);
    }
    int n = nt * 16 + col;
    float bs = bias1[n];
#pragma unroll
    for (int v = 0; v < 8; ++v) {
        int batch = wave * 16 + v + 8 * g;
        hidden[(size_t)batch * 512 + n] = lrelu(c[v] + bs);
    }
}

__global__ void dec2_kernel(const float* __restrict__ hidden, const float* __restrict__ w2,
                            const float* __restrict__ b2, float* __restrict__ out) {
    int b = threadIdx.x;  // 128 threads
    float acc = b2[0];
    for (int c = 0; c < 512; ++c) acc += hidden[(size_t)b * 512 + c] * w2[c];
    out[b] = acc;
}

// ---------------------------------------------------------------------------
extern "C" void kernel_launch(void* const* d_in, const int* in_sizes, int n_in,
                              void* d_out, int out_size, void* d_ws, size_t ws_size,
                              hipStream_t stream) {
    const float* pc_voxel  = (const float*)d_in[0];
    const float* global_pc = (const float*)d_in[1];
    const float* local_pc  = (const float*)d_in[2];
    const float* query     = (const float*)d_in[3];
    const float* voxel     = (const float*)d_in[4];
    const float* ge_w1 = (const float*)d_in[5];   const float* ge_b1 = (const float*)d_in[6];
    const float* ge_w2 = (const float*)d_in[7];   const float* ge_b2 = (const float*)d_in[8];
    // ge_w3/ge_b3 (d_in[9..10]) feed only global_c, which is dead code.
    const float* le_w1 = (const float*)d_in[11];  const float* le_b1 = (const float*)d_in[12];
    const float* le_w2 = (const float*)d_in[13];  const float* le_b2 = (const float*)d_in[14];
    const float* pos_w = (const float*)d_in[15];  const float* pos_b = (const float*)d_in[16];
    const float* bn_g  = (const float*)d_in[17];  const float* bn_b  = (const float*)d_in[18];
    const float* ap_w  = (const float*)d_in[19];  const float* ap_b  = (const float*)d_in[20];
    const float* as_w  = (const float*)d_in[21];  const float* as_b  = (const float*)d_in[22];
    const float* gcv_w1 = (const float*)d_in[23]; const float* gcv_b1 = (const float*)d_in[24];
    const float* gcv_w2 = (const float*)d_in[25]; const float* gcv_b2 = (const float*)d_in[26];
    const float* fcv_w1 = (const float*)d_in[27]; const float* fcv_b1 = (const float*)d_in[28];
    const float* fcv_w2 = (const float*)d_in[29]; const float* fcv_b2 = (const float*)d_in[30];
    const float* dec_w1 = (const float*)d_in[31]; const float* dec_b1 = (const float*)d_in[32];
    const float* dec_w2 = (const float*)d_in[33]; const float* dec_b2 = (const float*)d_in[34];

    char* ws = (char*)d_ws;
    size_t off = 0;
    auto carve = [&](size_t bytes) {
        void* p = ws + off;
        off = (off + bytes + 255) & ~(size_t)255;
        return p;
    };
    _Float16* decw1p = (_Float16*)carve((size_t)4608 * 512 * 2); // packed dec_w1
    _Float16* lew2p  = (_Float16*)carve((size_t)64 * 256 * 2);   // packed le_w2
    float*    featF  = (float*)carve((size_t)128 * 4608 * 4);    // decoder input, f32
    _Float16* featH  = (_Float16*)carve((size_t)128 * 4608 * 2); // decoder input, f16
    float*    nnpts  = (float*)carve((size_t)128 * 16 * 3 * 4);
    float*    knnpos = (float*)carve((size_t)128 * 16 * 3 * 4);
    float*    lepart = (float*)carve((size_t)1024 * 256 * 4);
    float*    hidden = (float*)carve((size_t)128 * 512 * 4);

    packw_kernel<<<(4608 * 512 + 255) / 256, 256, 0, stream>>>(dec_w1, decw1p, 4608, 512);
    packw_kernel<<<(64 * 256 + 255) / 256, 256, 0, stream>>>(le_w2, lew2p, 64, 256);

    knn_kernel<<<128, 256, 0, stream>>>(global_pc, query, nnpts, knnpos);
    att_kernel<<<128, 128, 0, stream>>>(nnpts, knnpos, query,
                                        ge_w1, ge_b1, ge_w2, ge_b2,
                                        pos_w, pos_b, bn_g, bn_b,
                                        ap_w, ap_b, as_w, as_b, featF);

    le_kernel<<<1024, 256, 0, stream>>>(local_pc, le_w1, le_b1, lew2p, le_b2, lepart);
    lered_kernel<<<128, 256, 0, stream>>>(lepart, featF);

    vox_kernel<<<256, 256, 0, stream>>>(pc_voxel, voxel,
                                        gcv_w1, gcv_b1, gcv_w2, gcv_b2,
                                        fcv_w1, fcv_b1, fcv_w2, fcv_b2, featF);

    featcvt_kernel<<<(128 * 4608 / 4 + 255) / 256, 256, 0, stream>>>(featF, featH);

    dec1_kernel<<<32, 256, 0, stream>>>(featH, decw1p, dec_b1, hidden);
    dec2_kernel<<<1, 128, 0, stream>>>(hidden, dec_w2, dec_b2, (float*)d_out);
}
